// RuleGNN_10806137716888
// MI455X (gfx1250) — compile-verified, matrix operand order checked by reading, not run on hardware
//
#include <hip/hip_runtime.h>

#define N_ENT   80000
#define N_EDGE  500000
#define N_REL   12
#define N_RULES 64
#define DD      64
#define N_LAYERS 2
#define BQ      256
#define N_CAND  500
#define N_ACT   4
#define LN_EPS  1e-5f
#define INV_SQRT_D 0.125f

typedef __attribute__((ext_vector_type(16))) _Float16 v16h;
typedef __attribute__((ext_vector_type(8)))  _Float16 v8h;
typedef __attribute__((ext_vector_type(8)))  float    v8f;

// ---------------- WMMA helpers (gfx1250, wave32) ----------------
__device__ __forceinline__ v8f wmma_f16(v16h a, v16h b, v8f c) {
  return __builtin_amdgcn_wmma_f32_16x16x32_f16(
      false, a, false, b, (short)0, c, false, false);
}

// A fragment from an f16 row: lane holds row m=lane&15, K pattern:
// halves [kk+hb, kk+hb+8) and [kk+16+hb, kk+16+hb+8), hb=(lane>>4)*8.
// Both chunks are contiguous & 16B aligned -> two global_load_b128.
__device__ __forceinline__ v16h load_a16(const _Float16* row, int kk, int half) {
  int hb = half * 8;
  union { v16h v; v8h h[2]; } u;
  u.h[0] = *(const v8h*)(row + kk + hb);
  u.h[1] = *(const v8h*)(row + kk + 16 + hb);
  return u.v;
}

// B fragment from pre-swizzled weights: 32 contiguous bytes per (frag,lane).
__device__ __forceinline__ v16h load_bswz(const _Float16* wm, int f, int lane) {
  return *(const v16h*)(wm + ((f * 32 + lane) << 4));
}

// orderable-uint encoding of float for atomicMax
__device__ __forceinline__ unsigned fkey(float f) {
  unsigned u = __float_as_uint(f);
  return (u & 0x80000000u) ? ~u : (u | 0x80000000u);
}
__device__ __forceinline__ float funkey(unsigned k) {
  return (k & 0x80000000u) ? __uint_as_float(k & 0x7fffffffu)
                           : __uint_as_float(~k);
}

// ---------------- prep kernels ----------------
__global__ void k_cvt(const float* __restrict__ s, _Float16* __restrict__ d, int n2) {
  int i = blockIdx.x * blockDim.x + threadIdx.x;
  if (i < n2) {
    union { unsigned u; _Float16 h[2]; } p;
    p.h[0] = (_Float16)s[2 * i];
    p.h[1] = (_Float16)s[2 * i + 1];
    ((unsigned*)d)[i] = p.u;
  }
}

// Swizzle all weight matrices into WMMA B-fragment order, once.
// Layout: 30 matrices of 64x64 (l*15 + {Wq,Wk1,Wk2,Wr0..11}), then s1_w (128x64).
__global__ void k_swz(const float* __restrict__ Wq_w, const float* __restrict__ Wk_w,
                      const float* __restrict__ W_r, const float* __restrict__ s1_w,
                      _Float16* __restrict__ wswz) {
  int t = blockIdx.x * blockDim.x + threadIdx.x;
  const float* W;
  _Float16* dst;
  int slot;
  if (t < 30 * 256) {
    int mat = t >> 8;
    slot = t & 255;
    int l = mat / 15, mi = mat % 15;
    if (mi == 0)      W = Wq_w + (size_t)l * 4096;
    else if (mi == 1) W = Wk_w + (size_t)l * 3 * 4096;
    else if (mi == 2) W = Wk_w + (size_t)l * 3 * 4096 + 4096;
    else              W = W_r  + (size_t)l * 12 * 4096 + (size_t)(mi - 3) * 4096;
    dst = wswz + (size_t)mat * 4096;
  } else if (t < 30 * 256 + 512) {
    slot = t - 30 * 256;          // 16 fragments (K=128)
    W = s1_w;
    dst = wswz + (size_t)30 * 4096;
  } else return;
  int f = slot >> 5, lane = slot & 31;
  int ks = f >> 2, nt = f & 3;
  int n = nt * 16 + (lane & 15);
  int hb = (lane >> 4) * 8;
  int kk = ks * 32;
  union { v16h v; _Float16 h[16]; } u;
#pragma unroll
  for (int j = 0; j < 8; ++j) {
    u.h[j]     = (_Float16)W[(kk + hb + j) * 64 + n];
    u.h[8 + j] = (_Float16)W[(kk + 16 + hb + j) * 64 + n];
  }
  *(v16h*)(dst + ((f * 32 + lane) << 4)) = u.v;
}

__global__ void k_params(const float* __restrict__ W_r_l,
                         const float* __restrict__ rule_emb_l,
                         const float* __restrict__ Wk_w_l,
                         const int* __restrict__ rule_ids,
                         _Float16* __restrict__ rel16,
                         float* __restrict__ rule_keys) {
  int t = threadIdx.x;
  if (t < N_REL * DD) {                 // W_r.mean(axis=-1) -> f16 A rows
    const float* p = W_r_l + (size_t)t * DD;
    float s = 0.f;
#pragma unroll 8
    for (int o = 0; o < DD; ++o) s += p[o];
    rel16[t] = (_Float16)(s * (1.0f / DD));
  } else if (t < N_REL * DD + N_ACT * DD) {
    int u = t - N_REL * DD;
    int a = u >> 6, i = u & 63;
    const float* re = rule_emb_l + (size_t)rule_ids[a] * DD;
    const float* wk = Wk_w_l + 2 * DD * DD;
    float s = 0.f;
#pragma unroll 8
    for (int j = 0; j < DD; ++j) s += re[j] * wk[j * DD + i];
    rule_keys[u] = s;
  }
}

__global__ void k_init(float* __restrict__ acc,
                       unsigned* __restrict__ segmax,
                       float* __restrict__ denom) {
  int i = blockIdx.x * blockDim.x + threadIdx.x;
  if (i < N_ENT * DD) acc[i] = 0.f;
  if (i < N_ENT * N_ACT) { segmax[i] = 0u; denom[i] = 0.f; }
}

__global__ void k_count_zero(unsigned* counts) {
  if (threadIdx.x < N_REL) counts[threadIdx.x] = 0u;
}
__global__ void k_hist(const int* __restrict__ et, unsigned* __restrict__ counts) {
  int e = blockIdx.x * blockDim.x + threadIdx.x;
  if (e < N_EDGE) atomicAdd(&counts[et[e]], 1u);
}
__global__ void k_scan(const unsigned* __restrict__ counts,
                       unsigned* __restrict__ reloff,
                       unsigned* __restrict__ tileoff,
                       unsigned* __restrict__ cursor) {
  unsigned ro = 0, to = 0;
  for (int r = 0; r < N_REL; ++r) {
    reloff[r] = ro; tileoff[r] = to; cursor[r] = ro;
    ro += counts[r];
    to += (counts[r] + 15u) >> 4;
  }
  reloff[N_REL] = ro; tileoff[N_REL] = to;
}
__global__ void k_scatter(const int* __restrict__ et,
                          unsigned* __restrict__ cursor,
                          int* __restrict__ sorted) {
  int e = blockIdx.x * blockDim.x + threadIdx.x;
  if (e < N_EDGE) {
    unsigned p = atomicAdd(&cursor[et[e]], 1u);
    sorted[p] = e;
  }
}

// ---------------- edge score kernel (WMMA, no LDS) ----------------
// 2 waves/block, 16 edges/wave, 31250 tiles exact.
__global__ void __launch_bounds__(64, 1)
k_edge_scores(const _Float16* __restrict__ h16,
              const int* __restrict__ src,
              const int* __restrict__ dst,
              const int* __restrict__ etype,
              const _Float16* __restrict__ wWq,
              const float* __restrict__ Wq_b,
              const _Float16* __restrict__ wWk1,
              const _Float16* __restrict__ wWk2,
              const float* __restrict__ Wk_b,
              const _Float16* __restrict__ rel16,
              const float* __restrict__ rule_keys,
              float* __restrict__ scores) {
  int wave = threadIdx.x >> 5;
  int lane = threadIdx.x & 31;
  int tile = blockIdx.x * 2 + wave;
  int m = lane & 15, half = lane >> 4;

  int e_m = tile * 16 + m;
  int dI = dst[e_m];
  int sI = src[e_m];
  int et = etype[e_m];
  const _Float16* hdst = h16 + (size_t)dI * DD;
  const _Float16* hsrc = h16 + (size_t)sI * DD;
  const _Float16* relr = rel16 + et * DD;

  float prod[8], rs0[8], rs1[8], rs2[8], rs3[8];
#pragma unroll
  for (int r = 0; r < 8; ++r) { prod[r] = rs0[r] = rs1[r] = rs2[r] = rs3[r] = 0.f; }

#pragma unroll
  for (int nt = 0; nt < 4; ++nt) {
    v8f q = {}; v8f k = {};
#pragma unroll
    for (int ks = 0; ks < 2; ++ks) {
      int kk = ks * 32, f = ks * 4 + nt;
      q = wmma_f16(load_a16(hdst, kk, half), load_bswz(wWq,  f, lane), q);
      k = wmma_f16(load_a16(hsrc, kk, half), load_bswz(wWk1, f, lane), k);
      k = wmma_f16(load_a16(relr, kk, half), load_bswz(wWk2, f, lane), k);
    }
    int n = nt * 16 + m;
    float qbias = Wq_b[n], kbias = Wk_b[n];
    float rk0 = rule_keys[0 * DD + n], rk1 = rule_keys[1 * DD + n];
    float rk2 = rule_keys[2 * DD + n], rk3 = rule_keys[3 * DD + n];
#pragma unroll
    for (int r = 0; r < 8; ++r) {
      float qv = q[r] + qbias;
      float kv = k[r] + kbias;
      prod[r] += qv * kv;
      rs0[r] += qv * rk0; rs1[r] += qv * rk1;
      rs2[r] += qv * rk2; rs3[r] += qv * rk3;
    }
  }
  // fold base score into each rule sum, then reduce 32 values instead of 40
#pragma unroll
  for (int r = 0; r < 8; ++r) {
    rs0[r] += prod[r]; rs1[r] += prod[r];
    rs2[r] += prod[r]; rs3[r] += prod[r];
  }
#pragma unroll
  for (int mk = 1; mk < 16; mk <<= 1) {
#pragma unroll
    for (int r = 0; r < 8; ++r) {
      rs0[r] += __shfl_xor(rs0[r], mk, 32);
      rs1[r] += __shfl_xor(rs1[r], mk, 32);
      rs2[r] += __shfl_xor(rs2[r], mk, 32);
      rs3[r] += __shfl_xor(rs3[r], mk, 32);
    }
  }
  if (m == 0) {            // lanes 0 and 16 write rows half*8 .. half*8+7
#pragma unroll
    for (int r = 0; r < 8; ++r) {
      int e = tile * 16 + half * 8 + r;
      float4 v;
      v.x = rs0[r] * INV_SQRT_D;
      v.y = rs1[r] * INV_SQRT_D;
      v.z = rs2[r] * INV_SQRT_D;
      v.w = rs3[r] * INV_SQRT_D;
      *(float4*)(scores + (size_t)e * 4) = v;
    }
  }
}

// ---------------- segment softmax ----------------
__global__ void k_segmax(const int* __restrict__ dst,
                         const float* __restrict__ scores,
                         unsigned* __restrict__ segmax) {
  int i = blockIdx.x * blockDim.x + threadIdx.x;
  if (i < N_EDGE * N_ACT) {
    int e = i >> 2, a = i & 3;
    atomicMax(&segmax[dst[e] * 4 + a], fkey(scores[i]));
  }
}
__global__ void k_exp(const int* __restrict__ dst,
                      float* __restrict__ scores,
                      const unsigned* __restrict__ segmax,
                      float* __restrict__ denom) {
  int i = blockIdx.x * blockDim.x + threadIdx.x;
  if (i < N_EDGE * N_ACT) {
    int e = i >> 2, a = i & 3;
    int d = dst[e] * 4 + a;
    float ex = __expf(scores[i] - funkey(segmax[d]));
    scores[i] = ex;
    atomicAdd(&denom[d], ex);
  }
}
__global__ void k_attnmean(const int* __restrict__ dst,
                           const float* __restrict__ ex,
                           const float* __restrict__ denom,
                           float* __restrict__ attn) {
  int e = blockIdx.x * blockDim.x + threadIdx.x;
  if (e < N_EDGE) {
    int d = dst[e] * 4;
    float s = ex[e * 4 + 0] / denom[d + 0] + ex[e * 4 + 1] / denom[d + 1] +
              ex[e * 4 + 2] / denom[d + 2] + ex[e * 4 + 3] / denom[d + 3];
    attn[e] = s * 0.25f;
  }
}

// ---------------- message kernel (WMMA over relation-bucketed edges) --------
__global__ void __launch_bounds__(64, 1)
k_msg(const _Float16* __restrict__ h16,
      const int* __restrict__ src,
      const int* __restrict__ dst,
      const int* __restrict__ sorted,
      const unsigned* __restrict__ reloff,
      const unsigned* __restrict__ tileoff,
      const _Float16* __restrict__ wWr,   // 12 swizzled matrices
      const float* __restrict__ attn,
      float* __restrict__ acc) {
  int wave = threadIdx.x >> 5;
  int lane = threadIdx.x & 31;
  int tile = blockIdx.x * 2 + wave;
  if ((unsigned)tile >= tileoff[N_REL]) return;

  int r = 0;
  while (r + 1 < N_REL && (unsigned)tile >= tileoff[r + 1]) ++r;
  unsigned base = reloff[r] + ((unsigned)tile - tileoff[r]) * 16u;
  int cnt = (int)min(16u, reloff[r + 1] - base);

  int m = lane & 15, half = lane >> 4;
  int sidx = sorted[base + (unsigned)((m < cnt) ? m : 0)];
  int sI = src[sidx];
  int dI = dst[sidx];
  float am = (m < cnt) ? attn[sidx] : 0.f;
  const _Float16* hrow = h16 + (size_t)sI * DD;
  const _Float16* W = wWr + (size_t)r * 4096;

  int   dj[8];
  float aj[8];
#pragma unroll
  for (int rr = 0; rr < 8; ++rr) {
    int row = rr + half * 8;
    dj[rr] = __shfl(dI, row, 32);
    aj[rr] = __shfl(am, row, 32);
  }

#pragma unroll
  for (int nt = 0; nt < 4; ++nt) {
    v8f c = {};
#pragma unroll
    for (int ks = 0; ks < 2; ++ks)
      c = wmma_f16(load_a16(hrow, ks * 32, half), load_bswz(W, ks * 4 + nt, lane), c);
    int n = nt * 16 + m;
#pragma unroll
    for (int rr = 0; rr < 8; ++rr) {
      int row = rr + half * 8;
      if (row < cnt) atomicAdd(&acc[(size_t)dj[rr] * DD + n], c[rr] * aj[rr]);
    }
  }
}

// ---------------- bias + LayerNorm + ReLU -> f16 out ----------------
__global__ void k_ln(const float* __restrict__ acc,
                     const float* __restrict__ cbias,
                     const float* __restrict__ g,
                     const float* __restrict__ b,
                     _Float16* __restrict__ h16out) {
  int wave = threadIdx.x >> 5;
  int lane = threadIdx.x & 31;
  int node = blockIdx.x * 4 + wave;
  if (node >= N_ENT) return;
  const float* row = acc + (size_t)node * DD;
  int j0 = 2 * lane, j1 = 2 * lane + 1;
  float x0 = row[j0] + cbias[j0];
  float x1 = row[j1] + cbias[j1];
  float s = x0 + x1, s2 = x0 * x0 + x1 * x1;
#pragma unroll
  for (int mk = 1; mk < 32; mk <<= 1) {
    s  += __shfl_xor(s, mk, 32);
    s2 += __shfl_xor(s2, mk, 32);
  }
  float mu = s * (1.0f / DD);
  float var = s2 * (1.0f / DD) - mu * mu;
  float inv = rsqrtf(var + LN_EPS);
  float y0 = fmaxf((x0 - mu) * inv * g[j0] + b[j0], 0.f);
  float y1 = fmaxf((x1 - mu) * inv * g[j1] + b[j1], 0.f);
  union { unsigned u; _Float16 h[2]; } p;
  p.h[0] = (_Float16)y0; p.h[1] = (_Float16)y1;
  ((unsigned*)(h16out + (size_t)node * DD))[lane] = p.u;
}

// ---------------- final scorer (WMMA, no LDS) ----------------
__global__ void __launch_bounds__(64, 1)
k_score(const _Float16* __restrict__ h16,
        const int* __restrict__ queries,
        const int* __restrict__ candidates,
        const _Float16* __restrict__ wS1,  // 16-frag swizzled 128x64
        const float* __restrict__ s1_b,
        const float* __restrict__ s2_w,
        const float* __restrict__ s2_b,
        float* __restrict__ out) {
  int wave = threadIdx.x >> 5;
  int lane = threadIdx.x & 31;
  int tile = blockIdx.x * 2 + wave;     // 8192 tiles exact
  int bq = tile >> 5;
  int ct = tile & 31;
  int m = lane & 15, half = lane >> 4;

  int cand0 = ct * 16;
  int ci = min(cand0 + m, N_CAND - 1);
  int head = queries[bq * 2 + 0];
  int tailI = candidates[bq * N_CAND + ci];
  const _Float16* hh = h16 + (size_t)head * DD;
  const _Float16* ht = h16 + (size_t)tailI * DD;

  float sprod[8];
#pragma unroll
  for (int rr = 0; rr < 8; ++rr) sprod[rr] = 0.f;

#pragma unroll
  for (int nt = 0; nt < 4; ++nt) {
    v8f c = {};
#pragma unroll
    for (int ks = 0; ks < 4; ++ks) {
      int kk = ks * 32;
      const _Float16* rowp = (kk < DD) ? (hh + kk) : (ht + (kk - DD));
      c = wmma_f16(load_a16(rowp, 0, half), load_bswz(wS1, ks * 4 + nt, lane), c);
    }
    int n = nt * 16 + m;
    float bias = s1_b[n], w2 = s2_w[n];
#pragma unroll
    for (int rr = 0; rr < 8; ++rr)
      sprod[rr] += fmaxf(c[rr] + bias, 0.f) * w2;
  }
#pragma unroll
  for (int mk = 1; mk < 16; mk <<= 1)
#pragma unroll
    for (int rr = 0; rr < 8; ++rr) sprod[rr] += __shfl_xor(sprod[rr], mk, 32);

  if (m == 0) {
    float s2b = s2_b[0];
#pragma unroll
    for (int rr = 0; rr < 8; ++rr) {
      int cand = cand0 + rr + half * 8;
      if (cand < N_CAND) out[bq * N_CAND + cand] = sprod[rr] + s2b;
    }
  }
}

// ---------------- launcher ----------------
extern "C" void kernel_launch(void* const* d_in, const int* in_sizes, int n_in,
                              void* d_out, int out_size, void* d_ws, size_t ws_size,
                              hipStream_t stream) {
  const int*   queries    = (const int*)d_in[0];
  const int*   edge_index = (const int*)d_in[1];
  const int*   edge_type  = (const int*)d_in[2];
  const int*   rule_ids   = (const int*)d_in[3];
  const int*   candidates = (const int*)d_in[4];
  const float* entity_emb = (const float*)d_in[5];
  const float* W_r        = (const float*)d_in[6];
  const float* Wq_w       = (const float*)d_in[7];
  const float* Wq_b       = (const float*)d_in[8];
  const float* Wk_w       = (const float*)d_in[9];
  const float* Wk_b       = (const float*)d_in[10];
  const float* rule_emb   = (const float*)d_in[11];
  const float* conv_bias  = (const float*)d_in[12];
  const float* ln_g       = (const float*)d_in[13];
  const float* ln_b       = (const float*)d_in[14];
  const float* s1_w       = (const float*)d_in[15];
  const float* s1_b       = (const float*)d_in[16];
  const float* s2_w       = (const float*)d_in[17];
  const float* s2_b       = (const float*)d_in[18];
  float* out = (float*)d_out;

  const int* src = edge_index;
  const int* dst = edge_index + N_EDGE;

  // workspace carve
  float*    accF    = (float*)d_ws;                          // N_ENT*DD
  float*    scores  = accF + (size_t)N_ENT * DD;             // N_EDGE*4
  unsigned* segmax  = (unsigned*)(scores + (size_t)N_EDGE * 4);
  float*    denom   = (float*)(segmax + (size_t)N_ENT * 4);
  float*    attn    = denom + (size_t)N_ENT * 4;             // N_EDGE
  float*    rkeys   = attn + N_EDGE;                         // 256
  int*      sorted  = (int*)(rkeys + N_ACT * DD);            // N_EDGE
  unsigned* counts  = (unsigned*)(sorted + N_EDGE);          // 12
  unsigned* reloff  = counts + N_REL;                        // 13
  unsigned* tileoff = reloff + (N_REL + 1);                  // 13
  unsigned* cursor  = tileoff + (N_REL + 1);                 // 12 (+pad 14 -> 64 total)
  _Float16* h16A    = (_Float16*)(cursor + N_REL + 14);      // N_ENT*DD halves
  _Float16* h16B    = h16A + (size_t)N_ENT * DD;
  _Float16* wswz    = h16B + (size_t)N_ENT * DD;             // 30*4096 + 8192 halves
  _Float16* rel16   = wswz + (size_t)30 * 4096 + 8192;       // 768 halves

  // one-time prep
  k_cvt<<<(N_ENT * DD / 2 + 255) / 256, 256, 0, stream>>>(entity_emb, h16A, N_ENT * DD / 2);
  k_swz<<<32, 256, 0, stream>>>(Wq_w, Wk_w, W_r, s1_w, wswz);
  k_count_zero<<<1, 64, 0, stream>>>(counts);
  k_hist<<<(N_EDGE + 255) / 256, 256, 0, stream>>>(edge_type, counts);
  k_scan<<<1, 1, 0, stream>>>(counts, reloff, tileoff, cursor);
  k_scatter<<<(N_EDGE + 255) / 256, 256, 0, stream>>>(edge_type, cursor, sorted);

  const _Float16* hin = h16A;
  _Float16* hout = h16B;
  for (int l = 0; l < N_LAYERS; ++l) {
    const _Float16* wWq  = wswz + (size_t)(l * 15 + 0) * 4096;
    const _Float16* wWk1 = wswz + (size_t)(l * 15 + 1) * 4096;
    const _Float16* wWk2 = wswz + (size_t)(l * 15 + 2) * 4096;
    const _Float16* wWr  = wswz + (size_t)(l * 15 + 3) * 4096;

    k_params<<<1, 1024, 0, stream>>>(W_r + (size_t)l * N_REL * DD * DD,
                                     rule_emb + (size_t)l * N_RULES * DD,
                                     Wk_w + (size_t)l * 3 * DD * DD,
                                     rule_ids, rel16, rkeys);
    k_init<<<(N_ENT * DD + 255) / 256, 256, 0, stream>>>(accF, segmax, denom);
    k_edge_scores<<<N_EDGE / 32, 64, 0, stream>>>(
        hin, src, dst, edge_type,
        wWq, Wq_b + (size_t)l * DD, wWk1, wWk2, Wk_b + (size_t)l * DD,
        rel16, rkeys, scores);
    k_segmax<<<(N_EDGE * 4 + 255) / 256, 256, 0, stream>>>(dst, scores, segmax);
    k_exp<<<(N_EDGE * 4 + 255) / 256, 256, 0, stream>>>(dst, scores, segmax, denom);
    k_attnmean<<<(N_EDGE + 255) / 256, 256, 0, stream>>>(dst, scores, denom, attn);
    k_msg<<<(N_EDGE / 16 + N_REL + 1) / 2, 64, 0, stream>>>(
        hin, src, dst, sorted, reloff, tileoff, wWr, attn, accF);
    k_ln<<<(N_ENT + 3) / 4, 128, 0, stream>>>(accF, conv_bias + (size_t)l * DD,
                                              ln_g + (size_t)l * DD,
                                              ln_b + (size_t)l * DD,
                                              hout);
    hin = hout;
    hout = h16A;   // layer 2 writes back into A
  }

  const _Float16* wS1 = wswz + (size_t)30 * 4096;
  k_score<<<(BQ * 32) / 2, 64, 0, stream>>>(h16A, queries, candidates,
                                            wS1, s1_b, s2_w, s2_b, out);
}
